// GAT_19782619365933
// MI455X (gfx1250) — matrix operand order repaired and checked by reading, not compiled, not run on hardware
//
#include <hip/hip_runtime.h>
#include <hip/hip_bf16.h>

typedef __attribute__((ext_vector_type(16))) _Float16 v16h;
typedef __attribute__((ext_vector_type(8)))  _Float16 v8h;
typedef __attribute__((ext_vector_type(4)))  _Float16 v4h;
typedef __attribute__((ext_vector_type(8)))  float    v8f;
typedef __attribute__((ext_vector_type(4)))  float    v4f;

#define FEAT 128
#define H1   8
#define C1   8
#define HID  64     // H1*C1
#define NCLS 20
#define NC2  32     // NCLS padded to 2x16 for WMMA

union V16 { v16h v; v8h h[2]; };

// ---------------------------------------------------------------- utilities

__global__ __launch_bounds__(256) void fill_f32(float* p, float v, int n) {
    int i = blockIdx.x * 256 + threadIdx.x;
    if (i < n) p[i] = v;
}

__device__ __forceinline__ void atomicMaxFloat(float* addr, float val) {
    // monotone bit-pattern trick; valid because init is -inf
    if (val >= 0.f) atomicMax((int*)addr, __float_as_int(val));
    else            atomicMin((unsigned int*)addr, __float_as_uint(val));
}

__device__ __forceinline__ float lrelu02(float x) { return x > 0.f ? x : 0.2f * x; }

// ---------------------------------------------------------------- stage 0: gathers / converts

// one thread converts 4 consecutive feats: float4 read -> 4xf16 (8B) write
__global__ __launch_bounds__(256) void embed_f16(const int* __restrict__ ids,
                                                 const float* __restrict__ emb,
                                                 _Float16* __restrict__ xh, int n) {
    int i = blockIdx.x * 256 + threadIdx.x;          // over n*(FEAT/4)
    if (i >= n * (FEAT / 4)) return;
    int node = i >> 5, q = (i & 31) << 2;
    v4f src = *(const v4f*)(emb + (long)ids[node] * FEAT + q);
    v4h dst;
#pragma unroll
    for (int t = 0; t < 4; ++t) dst[t] = (_Float16)src[t];
    *(v4h*)(xh + (long)node * FEAT + q) = dst;
}

// pack W1 into per-lane WMMA B-fragment order:
// w1p[(((nt*4)+c)*32 + lane)*16 + j] = W1[(c*32 + (j>=8)*16 + (lane>=16)*8 + (j&7)) * HID + nt*16 + (lane&15)]
__global__ __launch_bounds__(256) void pack_w1(const float* __restrict__ W1, _Float16* __restrict__ w1p) {
    int i = blockIdx.x * 256 + threadIdx.x;          // 4*4*32*16 = 8192
    if (i >= 4 * 4 * 32 * 16) return;
    int j    = i & 15;
    int lane = (i >> 4) & 31;
    int c    = (i >> 9) & 3;
    int nt   = i >> 11;
    int k    = c * 32 + ((j >> 3) << 4) + ((lane >> 4) << 3) + (j & 7);
    int nn   = nt * 16 + (lane & 15);
    w1p[i] = (_Float16)W1[k * HID + nn];
}

// same for W2 (64x20, n padded to 32)
__global__ __launch_bounds__(256) void pack_w2(const float* __restrict__ W2, _Float16* __restrict__ w2p) {
    int i = blockIdx.x * 256 + threadIdx.x;          // 2*2*32*16 = 2048
    if (i >= 2 * 2 * 32 * 16) return;
    int j    = i & 15;
    int lane = (i >> 4) & 31;
    int c    = (i >> 9) & 1;
    int nt   = i >> 10;
    int k    = c * 32 + ((j >> 3) << 4) + ((lane >> 4) << 3) + (j & 7);
    int nn   = nt * 16 + (lane & 15);
    w2p[i] = (_Float16)(nn < NCLS ? W2[k * NCLS + nn] : 0.f);
}

// ---------------------------------------------------------------- WMMA GEMM 1: xp = x @ W1  [n x 64]
// one wave computes a 16x64 stripe: A loaded once per k-chunk, 4 accumulators

__global__ __launch_bounds__(256) void gemm1_wmma(const _Float16* __restrict__ xh,
                                                  const _Float16* __restrict__ w1p,
                                                  float* __restrict__ xp, int n) {
    int wave = threadIdx.x >> 5;
    int lane = threadIdx.x & 31;
    int mt   = blockIdx.x * 8 + wave;
    int mtiles = (n + 15) >> 4;
    if (mt >= mtiles) return;

    int m   = lane & 15;
    int hi  = lane >> 4;                   // 0 or 1
    int row = mt * 16 + m;
    int rowc = row < n ? row : n - 1;      // clamp for tail loads

    v8f acc[4] = {{}, {}, {}, {}};
#pragma unroll
    for (int c = 0; c < 4; ++c) {
        // A fragment: j=0..7 -> k = k0 + hi*8 + j ; j=8..15 -> k = k0 + 16 + hi*8 + (j-8)
        long abase = (long)rowc * FEAT + c * 32 + (hi << 3);
        V16 A;
        A.h[0] = *(const v8h*)(xh + abase);
        A.h[1] = *(const v8h*)(xh + abase + 16);
#pragma unroll
        for (int nt = 0; nt < 4; ++nt) {
            v16h b = *(const v16h*)(w1p + ((((nt << 2) + c) << 5) + lane) * 16);
            acc[nt] = __builtin_amdgcn_wmma_f32_16x16x32_f16(false, A.v, false, b, (short)0, acc[nt], false, false);
        }
    }
#pragma unroll
    for (int nt = 0; nt < 4; ++nt)
#pragma unroll
        for (int r = 0; r < 8; ++r) {
            int orow = mt * 16 + hi * 8 + r;       // D layout: lanes 16-31 hold M=8..15
            if (orow < n) xp[(long)orow * HID + nt * 16 + m] = acc[nt][r];
        }
}

// ---------------------------------------------------------------- attention scalars, layer 1

__global__ __launch_bounds__(256) void attvec1(const float* __restrict__ xp,
                                               const float* __restrict__ a_src,
                                               const float* __restrict__ a_dst,
                                               float* __restrict__ asrc, float* __restrict__ adst, int n) {
    int i = blockIdx.x * 256 + threadIdx.x;          // over n*H1
    if (i >= n * H1) return;
    int node = i >> 3, h = i & 7;
    const float* v  = &xp[(long)node * HID + h * C1];
    float s = 0.f, d = 0.f;
#pragma unroll
    for (int c = 0; c < C1; ++c) { s += v[c] * a_src[h * C1 + c]; d += v[c] * a_dst[h * C1 + c]; }
    asrc[i] = s; adst[i] = d;
}

// ---------------------------------------------------------------- layer-1 edge passes (thread per edge*head)

__global__ __launch_bounds__(256) void att1_max(const int* __restrict__ ei,
                                                const float* __restrict__ asrc, const float* __restrict__ adst,
                                                float* __restrict__ m1, int nE, int n) {
    int t = blockIdx.x * 256 + threadIdx.x;
    int total = nE + n;
    int eid = t >> 3, h = t & 7;
    if (eid >= total) return;
    int s = eid < nE ? ei[eid] : eid - nE;
    int d = eid < nE ? ei[nE + eid] : eid - nE;
    float e = lrelu02(asrc[s * H1 + h] + adst[d * H1 + h]);
    atomicMaxFloat(&m1[d * H1 + h], e);
}

__global__ __launch_bounds__(256) void att1_sum(const int* __restrict__ ei,
                                                const float* __restrict__ asrc, const float* __restrict__ adst,
                                                const float* __restrict__ m1, float* __restrict__ s1,
                                                int nE, int n) {
    int t = blockIdx.x * 256 + threadIdx.x;
    int total = nE + n;
    int eid = t >> 3, h = t & 7;
    if (eid >= total) return;
    int s = eid < nE ? ei[eid] : eid - nE;
    int d = eid < nE ? ei[nE + eid] : eid - nE;
    float e = lrelu02(asrc[s * H1 + h] + adst[d * H1 + h]);
    atomicAdd(&s1[d * H1 + h], __expf(e - m1[d * H1 + h]));
}

__global__ __launch_bounds__(256) void att1_agg(const int* __restrict__ ei,
                                                const float* __restrict__ asrc, const float* __restrict__ adst,
                                                const float* __restrict__ m1, const float* __restrict__ s1,
                                                const float* __restrict__ xp, float* __restrict__ agg,
                                                int nE, int n) {
    int t = blockIdx.x * 256 + threadIdx.x;
    int total = nE + n;
    int eid = t >> 3, h = t & 7;
    if (eid >= total) return;
    int s = eid < nE ? ei[eid] : eid - nE;
    int d = eid < nE ? ei[nE + eid] : eid - nE;
    int dh = d * H1 + h;
    float e = lrelu02(asrc[s * H1 + h] + adst[dh]);
    float alpha = __expf(e - m1[dh]) / s1[dh];
    const float* v = &xp[(long)s * HID + h * C1];
    float* o = &agg[(long)d * HID + h * C1];
#pragma unroll
    for (int c = 0; c < C1; ++c) atomicAdd(&o[c], v[c] * alpha);
}

// ---------------------------------------------------------------- h = elu(agg + b1), stored f16 for GEMM2

__global__ __launch_bounds__(256) void finalize1(const float* __restrict__ agg,
                                                 const float* __restrict__ b1,
                                                 _Float16* __restrict__ hh, int n) {
    int i = blockIdx.x * 256 + threadIdx.x;          // over n*HID
    if (i >= n * HID) return;
    float v = agg[i] + b1[i & (HID - 1)];
    v = v > 0.f ? v : __expf(v) - 1.f;
    hh[i] = (_Float16)v;
}

// ---------------------------------------------------------------- WMMA GEMM 2: xp2 = h @ W2  [n x 20]
// one wave computes a 16x32 stripe (cols 20..31 discarded)

__global__ __launch_bounds__(256) void gemm2_wmma(const _Float16* __restrict__ hh,
                                                  const _Float16* __restrict__ w2p,
                                                  float* __restrict__ xp2, int n) {
    int wave = threadIdx.x >> 5;
    int lane = threadIdx.x & 31;
    int mt   = blockIdx.x * 8 + wave;
    int mtiles = (n + 15) >> 4;
    if (mt >= mtiles) return;

    int m   = lane & 15;
    int hi  = lane >> 4;
    int row = mt * 16 + m;
    int rowc = row < n ? row : n - 1;

    v8f acc[2] = {{}, {}};
#pragma unroll
    for (int c = 0; c < 2; ++c) {          // K = 64
        long abase = (long)rowc * HID + c * 32 + (hi << 3);
        V16 A;
        A.h[0] = *(const v8h*)(hh + abase);
        A.h[1] = *(const v8h*)(hh + abase + 16);
#pragma unroll
        for (int nt = 0; nt < 2; ++nt) {
            v16h b = *(const v16h*)(w2p + ((((nt << 1) + c) << 5) + lane) * 16);
            acc[nt] = __builtin_amdgcn_wmma_f32_16x16x32_f16(false, A.v, false, b, (short)0, acc[nt], false, false);
        }
    }
#pragma unroll
    for (int nt = 0; nt < 2; ++nt) {
        int col = nt * 16 + m;
        if (col >= NCLS) continue;
#pragma unroll
        for (int r = 0; r < 8; ++r) {
            int orow = mt * 16 + hi * 8 + r;
            if (orow < n) xp2[(long)orow * NCLS + col] = acc[nt][r];
        }
    }
}

// ---------------------------------------------------------------- attention scalars, layer 2 (H=1)

__global__ __launch_bounds__(256) void attvec2(const float* __restrict__ xp2,
                                               const float* __restrict__ a_src, const float* __restrict__ a_dst,
                                               float* __restrict__ asrc, float* __restrict__ adst, int n) {
    int node = blockIdx.x * 256 + threadIdx.x;
    if (node >= n) return;
    const float* v = &xp2[(long)node * NCLS];
    float s = 0.f, d = 0.f;
#pragma unroll
    for (int c = 0; c < NCLS; ++c) { s += v[c] * a_src[c]; d += v[c] * a_dst[c]; }
    asrc[node] = s; adst[node] = d;
}

__global__ __launch_bounds__(256) void att2_max(const int* __restrict__ ei,
                                                const float* __restrict__ asrc, const float* __restrict__ adst,
                                                float* __restrict__ m2, int nE, int n) {
    int eid = blockIdx.x * 256 + threadIdx.x;
    if (eid >= nE + n) return;
    int s = eid < nE ? ei[eid] : eid - nE;
    int d = eid < nE ? ei[nE + eid] : eid - nE;
    atomicMaxFloat(&m2[d], lrelu02(asrc[s] + adst[d]));
}

__global__ __launch_bounds__(256) void att2_sum(const int* __restrict__ ei,
                                                const float* __restrict__ asrc, const float* __restrict__ adst,
                                                const float* __restrict__ m2, float* __restrict__ s2,
                                                int nE, int n) {
    int eid = blockIdx.x * 256 + threadIdx.x;
    if (eid >= nE + n) return;
    int s = eid < nE ? ei[eid] : eid - nE;
    int d = eid < nE ? ei[nE + eid] : eid - nE;
    float e = lrelu02(asrc[s] + adst[d]);
    atomicAdd(&s2[d], __expf(e - m2[d]));
}

__global__ __launch_bounds__(256) void att2_agg(const int* __restrict__ ei,
                                                const float* __restrict__ asrc, const float* __restrict__ adst,
                                                const float* __restrict__ m2, const float* __restrict__ s2,
                                                const float* __restrict__ xp2, float* __restrict__ agg2,
                                                int nE, int n) {
    int eid = blockIdx.x * 256 + threadIdx.x;
    if (eid >= nE + n) return;
    int s = eid < nE ? ei[eid] : eid - nE;
    int d = eid < nE ? ei[nE + eid] : eid - nE;
    float e = lrelu02(asrc[s] + adst[d]);
    float alpha = __expf(e - m2[d]) / s2[d];
    const float* v = &xp2[(long)s * NCLS];
    float* o = &agg2[(long)d * NCLS];
#pragma unroll
    for (int c = 0; c < NCLS; ++c) atomicAdd(&o[c], v[c] * alpha);
}

// ---------------------------------------------------------------- log_softmax(agg2 + b2)

__global__ __launch_bounds__(256) void final_lsm(const float* __restrict__ agg2,
                                                 const float* __restrict__ b2,
                                                 float* __restrict__ out, int n) {
    int node = blockIdx.x * 256 + threadIdx.x;
    if (node >= n) return;
    float o[NCLS], mx = -1e30f;
#pragma unroll
    for (int c = 0; c < NCLS; ++c) { o[c] = agg2[(long)node * NCLS + c] + b2[c]; mx = fmaxf(mx, o[c]); }
    float se = 0.f;
#pragma unroll
    for (int c = 0; c < NCLS; ++c) se += __expf(o[c] - mx);
    float lse = mx + __logf(se);
#pragma unroll
    for (int c = 0; c < NCLS; ++c) out[(long)node * NCLS + c] = o[c] - lse;
}

// ---------------------------------------------------------------- launch

static inline size_t alignup(size_t x) { return (x + 255) & ~(size_t)255; }

extern "C" void kernel_launch(void* const* d_in, const int* in_sizes, int n_in,
                              void* d_out, int out_size, void* d_ws, size_t ws_size,
                              hipStream_t stream) {
    const int*   x_ids  = (const int*)d_in[0];
    const int*   ei     = (const int*)d_in[1];
    const float* emb    = (const float*)d_in[2];
    const float* W1     = (const float*)d_in[3];
    const float* a_src1 = (const float*)d_in[4];
    const float* a_dst1 = (const float*)d_in[5];
    const float* b1     = (const float*)d_in[6];
    const float* W2     = (const float*)d_in[7];
    const float* a_src2 = (const float*)d_in[8];
    const float* a_dst2 = (const float*)d_in[9];
    const float* b2     = (const float*)d_in[10];
    float* out = (float*)d_out;

    const int n  = in_sizes[0];            // 50000
    const int nE = in_sizes[1] / 2;        // 1.6M
    const int nEt = nE + n;                // with self-loops

    // workspace carve-up
    char* p = (char*)d_ws;
    _Float16* xh   = (_Float16*)p;  p += alignup((size_t)n * FEAT * 2);
    _Float16* w1p  = (_Float16*)p;  p += alignup((size_t)FEAT * HID * 2);
    _Float16* w2p  = (_Float16*)p;  p += alignup((size_t)HID * NC2 * 2);
    float* xp      = (float*)p;     p += alignup((size_t)n * HID * 4);
    float* asrc1   = (float*)p;     p += alignup((size_t)n * H1 * 4);
    float* adst1   = (float*)p;     p += alignup((size_t)n * H1 * 4);
    float* m1      = (float*)p;     p += alignup((size_t)n * H1 * 4);
    float* s1      = (float*)p;     p += alignup((size_t)n * H1 * 4);
    float* agg1    = (float*)p;     p += alignup((size_t)n * HID * 4);
    _Float16* hh   = (_Float16*)p;  p += alignup((size_t)n * HID * 2);
    float* xp2     = (float*)p;     p += alignup((size_t)n * NCLS * 4);
    float* asrc2   = (float*)p;     p += alignup((size_t)n * 4);
    float* adst2   = (float*)p;     p += alignup((size_t)n * 4);
    float* m2      = (float*)p;     p += alignup((size_t)n * 4);
    float* s2      = (float*)p;     p += alignup((size_t)n * 4);
    float* agg2    = (float*)p;     p += alignup((size_t)n * NCLS * 4);

    auto cdiv = [](int a, int b) { return (a + b - 1) / b; };
    const float NEG_INF = -__builtin_huge_valf();

    // stage 0: converts + gathers
    embed_f16<<<cdiv(n * (FEAT / 4), 256), 256, 0, stream>>>(x_ids, emb, xh, n);
    pack_w1<<<cdiv(FEAT * HID, 256), 256, 0, stream>>>(W1, w1p);
    pack_w2<<<cdiv(HID * NC2, 256), 256, 0, stream>>>(W2, w2p);

    // init segment state
    fill_f32<<<cdiv(n * H1, 256), 256, 0, stream>>>(m1, NEG_INF, n * H1);
    fill_f32<<<cdiv(n * H1, 256), 256, 0, stream>>>(s1, 0.f, n * H1);
    fill_f32<<<cdiv(n * HID, 256), 256, 0, stream>>>(agg1, 0.f, n * HID);
    fill_f32<<<cdiv(n, 256), 256, 0, stream>>>(m2, NEG_INF, n);
    fill_f32<<<cdiv(n, 256), 256, 0, stream>>>(s2, 0.f, n);
    fill_f32<<<cdiv(n * NCLS, 256), 256, 0, stream>>>(agg2, 0.f, n * NCLS);

    // layer 1
    int mtiles = cdiv(n, 16);
    gemm1_wmma<<<cdiv(mtiles, 8), 256, 0, stream>>>(xh, w1p, xp, n);
    attvec1<<<cdiv(n * H1, 256), 256, 0, stream>>>(xp, a_src1, a_dst1, asrc1, adst1, n);
    att1_max<<<cdiv(nEt * H1, 256), 256, 0, stream>>>(ei, asrc1, adst1, m1, nE, n);
    att1_sum<<<cdiv(nEt * H1, 256), 256, 0, stream>>>(ei, asrc1, adst1, m1, s1, nE, n);
    att1_agg<<<cdiv(nEt * H1, 256), 256, 0, stream>>>(ei, asrc1, adst1, m1, s1, xp, agg1, nE, n);
    finalize1<<<cdiv(n * HID, 256), 256, 0, stream>>>(agg1, b1, hh, n);

    // layer 2
    gemm2_wmma<<<cdiv(mtiles, 8), 256, 0, stream>>>(hh, w2p, xp2, n);
    attvec2<<<cdiv(n, 256), 256, 0, stream>>>(xp2, a_src2, a_dst2, asrc2, adst2, n);
    att2_max<<<cdiv(nEt, 256), 256, 0, stream>>>(ei, asrc2, adst2, m2, nE, n);
    att2_sum<<<cdiv(nEt, 256), 256, 0, stream>>>(ei, asrc2, adst2, m2, s2, nE, n);
    att2_agg<<<cdiv(nEt, 256), 256, 0, stream>>>(ei, asrc2, adst2, m2, s2, xp2, agg2, nE, n);

    // output
    final_lsm<<<cdiv(n, 256), 256, 0, stream>>>(agg2, b2, out, n);
}
